// Bidirectional_GRU_Decoder_47399259079348
// MI455X (gfx1250) — matrix operand order repaired and checked by reading, not compiled, not run on hardware
//
#include <hip/hip_runtime.h>
#include <hip/hip_bf16.h>

#define B_  64
#define S_  512
#define D_  512
#define H_  512
#define L_  2
#define G3H 1536   /* 3*H */
#define KDIM 512

typedef __bf16 bf16_t;
typedef __attribute__((ext_vector_type(16))) __bf16 v16bf;
typedef __attribute__((ext_vector_type(8)))  float  v8f;

// ---------------------------------------------------------------------------
// Descriptors for the per-step batched GEMMs:  C(64x1536) = A(64x512) * B^T
// B is stored row-major as (N=1536 rows, K=512 cols)  (i.e. W, computing x@W^T)
// ---------------------------------------------------------------------------
struct GemmDesc {
  const bf16_t* A;   // 64 rows, row stride As (elements)
  long          As;
  const bf16_t* Bm;  // 1536 rows x 512, row stride 512
  float*        C;   // 64 x 1536
};
struct GemmArgs { GemmDesc g[4]; };

// A-fragment: lane l<16 holds row M=l, K = kb..kb+7 (V0-3) and kb+16..kb+23
// (V4-7) with kb = (l>>4)*8  -> two b128 loads at +0 and +16 elements.
__device__ inline v16bf ld_frag_a(const bf16_t* p) {
  union { uint4 q[2]; v16bf v; } u;
  u.q[0] = *(const uint4*)(p);
  u.q[1] = *(const uint4*)(p + 16);
  return u.v;
}
// B-fragment: lane l holds column N=l%16, K = kb..kb+7 and kb+16..kb+23
// with kb = (l>>4)*8 -> two b128 loads at +0 and +16 elements.
__device__ inline v16bf ld_frag_b(const bf16_t* p) {
  union { uint4 q[2]; v16bf v; } u;
  u.q[0] = *(const uint4*)(p);
  u.q[1] = *(const uint4*)(p + 16);
  return u.v;
}

__device__ inline v8f wmma_bf16(v16bf a, v16bf b, v8f c) {
  return __builtin_amdgcn_wmma_f32_16x16x32_bf16(
      /*neg_a=*/false, a, /*neg_b=*/false, b,
      /*c_mod=*/(short)0, c, /*reuse_a=*/false, /*reuse_b=*/false);
}

// One wave computes the full M=64 column of one 16-wide N tile:
// 4 accumulators, B fragment loaded once per K step and reused 4x in registers.
__global__ __launch_bounds__(128)
void step_gemm_kernel(GemmArgs args) {
  const GemmDesc d = args.g[blockIdx.z];
  const int lane  = threadIdx.x & 31;
  const int wave  = threadIdx.x >> 5;
  const int ntile = blockIdx.x * 4 + wave;   // 0..95
  const int half  = lane >> 4;               // 0/1
  const int l16   = lane & 15;

  const bf16_t* ap = d.A  + (size_t)l16 * d.As + half * 8;
  const bf16_t* bp = d.Bm + (size_t)(ntile * 16 + l16) * KDIM + half * 8;
  const size_t mstep = (size_t)16 * d.As;    // advance one 16-row M tile

  v8f c0 = {}, c1 = {}, c2 = {}, c3 = {};
#pragma unroll
  for (int k0 = 0; k0 < KDIM; k0 += 32) {
    v16bf b  = ld_frag_b(bp); bp += 32;
    v16bf a0 = ld_frag_a(ap);
    v16bf a1 = ld_frag_a(ap + mstep);
    v16bf a2 = ld_frag_a(ap + 2 * mstep);
    v16bf a3 = ld_frag_a(ap + 3 * mstep);
    ap += 32;
    c0 = wmma_bf16(a0, b, c0);
    c1 = wmma_bf16(a1, b, c1);
    c2 = wmma_bf16(a2, b, c2);
    c3 = wmma_bf16(a3, b, c3);
  }

  // C/D layout: VGPR r, lanes 0-15 -> M = r, lanes 16-31 -> M = 8+r; N = l16
  float* cp = d.C + (size_t)(half * 8) * G3H + ntile * 16 + l16;
#pragma unroll
  for (int r = 0; r < 8; ++r) {
    cp[(size_t)(r +  0) * G3H] = c0[r];
    cp[(size_t)(r + 16) * G3H] = c1[r];
    cp[(size_t)(r + 32) * G3H] = c2[r];
    cp[(size_t)(r + 48) * G3H] = c3[r];
  }
}

// ---------------------------------------------------------------------------
// Fused GRU cell (both directions per launch via blockIdx.y)
// ---------------------------------------------------------------------------
struct PwDesc {
  const float* gi; const float* gh;
  const float* bih; const float* bhh;
  float* h; bf16_t* hbf;
  float* out;   // null for layer 0
  int    tpos;  // time index into out (already flipped for backward)
};
struct PwArgs { PwDesc d[2]; };

__global__ __launch_bounds__(256)
void gru_pointwise_kernel(PwArgs args) {
  const PwDesc d = args.d[blockIdx.y];
  const int idx = blockIdx.x * 256 + threadIdx.x;   // 0 .. B*H-1
  const int b = idx >> 9;
  const int n = idx & (H_ - 1);

  const float* gir = d.gi + (size_t)b * G3H;
  const float* ghr = d.gh + (size_t)b * G3H;

  float ir  = gir[n]          + d.bih[n];
  float iz  = gir[H_ + n]     + d.bih[H_ + n];
  float in_ = gir[2 * H_ + n] + d.bih[2 * H_ + n];
  float hr  = ghr[n]          + d.bhh[n];
  float hz  = ghr[H_ + n]     + d.bhh[H_ + n];
  float hn  = ghr[2 * H_ + n] + d.bhh[2 * H_ + n];

  float r  = 1.0f / (1.0f + __expf(-(ir + hr)));
  float z  = 1.0f / (1.0f + __expf(-(iz + hz)));
  float nn = tanhf(in_ + r * hn);
  float hp = d.h[idx];
  float hv = (1.0f - z) * nn + z * hp;

  d.h[idx]   = hv;
  d.hbf[idx] = (bf16_t)hv;
  if (d.out)
    d.out[((size_t)b * S_ + d.tpos) * (2 * H_) + n] = hv;
}

// ---------------------------------------------------------------------------
// Prep / epilogue kernels
// ---------------------------------------------------------------------------
__global__ void f32_to_bf16_kernel(const float* __restrict__ src,
                                   bf16_t* __restrict__ dst, int n) {
  int i = blockIdx.x * blockDim.x + threadIdx.x;
  int stride = gridDim.x * blockDim.x;
  for (; i < n; i += stride) dst[i] = (bf16_t)src[i];
}

__global__ void init_h_kernel(const float* __restrict__ enc,
                              float* hf, float* hb, bf16_t* hfbf, bf16_t* hbbf) {
  int i = blockIdx.x * blockDim.x + threadIdx.x;   // 0 .. L*B*H-1
  if (i >= L_ * B_ * H_) return;
  int lb = i >> 9;             // l*B + b
  int n  = i & (H_ - 1);
  float vf = enc[(size_t)lb * (2 * H_) + n];
  float vb = enc[(size_t)lb * (2 * H_) + H_ + n];
  hf[i] = vf;  hb[i] = vb;
  hfbf[i] = (bf16_t)vf;  hbbf[i] = (bf16_t)vb;
}

__global__ void final_h_kernel(const float* hf, const float* hb, float* outh) {
  int i = blockIdx.x * blockDim.x + threadIdx.x;
  if (i >= L_ * B_ * H_) return;
  int lb = i >> 9;
  int n  = i & (H_ - 1);
  outh[(size_t)lb * (2 * H_) + n]      = hf[i];
  outh[(size_t)lb * (2 * H_) + H_ + n] = hb[i];
}

// ---------------------------------------------------------------------------
extern "C" void kernel_launch(void* const* d_in, const int* in_sizes, int n_in,
                              void* d_out, int out_size, void* d_ws, size_t ws_size,
                              hipStream_t stream) {
  (void)in_sizes; (void)n_in; (void)out_size; (void)ws_size;

  const float* input = (const float*)d_in[0];
  const float* enc   = (const float*)d_in[1];
  const float* Wih_f = (const float*)d_in[2];
  const float* Whh_f = (const float*)d_in[3];
  const float* bih_f = (const float*)d_in[4];
  const float* bhh_f = (const float*)d_in[5];
  const float* Wih_b = (const float*)d_in[6];
  const float* Whh_b = (const float*)d_in[7];
  const float* bih_b = (const float*)d_in[8];
  const float* bhh_b = (const float*)d_in[9];

  float* out   = (float*)d_out;                        // (B,S,2H)
  float* out_h = out + (size_t)B_ * S_ * 2 * H_;       // (L,B,2H)

  // ---- workspace carve-up (~47 MB, all fully initialized each call) ----
  char* ws = (char*)d_ws;
  size_t off = 0;
  auto carve = [&](size_t bytes) -> char* {
    char* p = ws + off;
    off += (bytes + 255) & ~(size_t)255;
    return p;
  };
  const size_t NW = (size_t)L_ * G3H * D_;             // elems per weight tensor
  bf16_t* xbf  = (bf16_t*)carve((size_t)B_ * S_ * D_ * 2);
  bf16_t* wihf = (bf16_t*)carve(NW * 2);
  bf16_t* whhf = (bf16_t*)carve(NW * 2);
  bf16_t* wihb = (bf16_t*)carve(NW * 2);
  bf16_t* whhb = (bf16_t*)carve(NW * 2);
  float*  hf   = (float*) carve((size_t)L_ * B_ * H_ * 4);
  float*  hb   = (float*) carve((size_t)L_ * B_ * H_ * 4);
  bf16_t* hfbf = (bf16_t*)carve((size_t)L_ * B_ * H_ * 2);
  bf16_t* hbbf = (bf16_t*)carve((size_t)L_ * B_ * H_ * 2);
  float*  gi_f = (float*) carve((size_t)B_ * G3H * 4);
  float*  gh_f = (float*) carve((size_t)B_ * G3H * 4);
  float*  gi_b = (float*) carve((size_t)B_ * G3H * 4);
  float*  gh_b = (float*) carve((size_t)B_ * G3H * 4);

  // ---- one-time conversions to bf16 ----
  const int nx = B_ * S_ * D_;
  f32_to_bf16_kernel<<<1024, 256, 0, stream>>>(input, xbf, nx);
  f32_to_bf16_kernel<<<512, 256, 0, stream>>>(Wih_f, wihf, (int)NW);
  f32_to_bf16_kernel<<<512, 256, 0, stream>>>(Whh_f, whhf, (int)NW);
  f32_to_bf16_kernel<<<512, 256, 0, stream>>>(Wih_b, wihb, (int)NW);
  f32_to_bf16_kernel<<<512, 256, 0, stream>>>(Whh_b, whhb, (int)NW);
  init_h_kernel<<<(L_ * B_ * H_ + 255) / 256, 256, 0, stream>>>(enc, hf, hb, hfbf, hbbf);

  // 96 n-tiles (4 per block) x full-M waves x 4 GEMM descriptors
  const dim3 gblk(128), ggrid(24, 1, 4);
  const dim3 pblk(256), pgrid(B_ * H_ / 256, 2);
  const size_t BH = (size_t)B_ * H_;

  for (int t = 0; t < S_; ++t) {
    // -------- layer 0 (input from sequence; fwd at t, bwd at S-1-t) --------
    GemmArgs ga0;
    ga0.g[0] = { xbf + (size_t)t * D_,            (long)S_ * D_, wihf, gi_f };
    ga0.g[1] = { hfbf,                            (long)H_,      whhf, gh_f };
    ga0.g[2] = { xbf + (size_t)(S_ - 1 - t) * D_, (long)S_ * D_, wihb, gi_b };
    ga0.g[3] = { hbbf,                            (long)H_,      whhb, gh_b };
    step_gemm_kernel<<<ggrid, gblk, 0, stream>>>(ga0);

    PwArgs pa0;
    pa0.d[0] = { gi_f, gh_f, bih_f, bhh_f, hf, hfbf, nullptr, 0 };
    pa0.d[1] = { gi_b, gh_b, bih_b, bhh_b, hb, hbbf, nullptr, 0 };
    gru_pointwise_kernel<<<pgrid, pblk, 0, stream>>>(pa0);

    // -------- layer 1 (input = new layer-0 hidden state) --------
    GemmArgs ga1;
    ga1.g[0] = { hfbf,      (long)H_, wihf + NW / 2, gi_f };   // Wih_f[1]
    ga1.g[1] = { hfbf + BH, (long)H_, whhf + NW / 2, gh_f };   // Whh_f[1]
    ga1.g[2] = { hbbf,      (long)H_, wihb + NW / 2, gi_b };
    ga1.g[3] = { hbbf + BH, (long)H_, whhb + NW / 2, gh_b };
    step_gemm_kernel<<<ggrid, gblk, 0, stream>>>(ga1);

    PwArgs pa1;
    pa1.d[0] = { gi_f, gh_f, bih_f + G3H, bhh_f + G3H, hf + BH, hfbf + BH, out,      t          };
    pa1.d[1] = { gi_b, gh_b, bih_b + G3H, bhh_b + G3H, hb + BH, hbbf + BH, out + H_, S_ - 1 - t };
    gru_pointwise_kernel<<<pgrid, pblk, 0, stream>>>(pa1);
  }

  final_h_kernel<<<(L_ * B_ * H_ + 255) / 256, 256, 0, stream>>>(hf, hb, out_h);
}
